// VGRNN_82660940579224
// MI455X (gfx1250) — matrix operand order repaired and checked by reading, not compiled, not run on hardware
//
#include <hip/hip_runtime.h>
#include <math.h>

// ---------------- problem constants (match reference) ----------------
#define NN_   5000
#define EE_   160000
#define TT_   3
#define XD_   256
#define HD_   256
#define ZD_   128
#define EPSF  1e-10f

typedef __bf16 bf16_t;
typedef __attribute__((ext_vector_type(16))) __bf16 v16bf;
typedef __attribute__((ext_vector_type(8)))  __bf16 v8bf;
typedef __attribute__((ext_vector_type(8)))  float  v8f;

enum { ACT_NONE = 0, ACT_RELU = 1, ACT_SOFTPLUS = 2 };

__device__ __forceinline__ float softplusf_(float x) {
  // stable: log1p(exp(-|x|)) + max(x,0)
  return log1pf(expf(-fabsf(x))) + fmaxf(x, 0.f);
}

// A-fragment (16x32 bf16, ISA 7.12.2): lane half h owns K runs
// [h*8, h*8+8) and [16+h*8, 16+h*8+8) -> two contiguous 8-float runs.
__device__ __forceinline__ v16bf load_a_frag_f32(const float* __restrict__ Arow,
                                                 int kbase, int half) {
  const float4* p0 = (const float4*)(Arow + kbase + half * 8);
  const float4* p1 = (const float4*)(Arow + kbase + 16 + half * 8);
  const float4 x0 = p0[0], x1 = p0[1], y0 = p1[0], y1 = p1[1];
  v16bf a;
  a[0]  = (bf16_t)x0.x; a[1]  = (bf16_t)x0.y; a[2]  = (bf16_t)x0.z; a[3]  = (bf16_t)x0.w;
  a[4]  = (bf16_t)x1.x; a[5]  = (bf16_t)x1.y; a[6]  = (bf16_t)x1.z; a[7]  = (bf16_t)x1.w;
  a[8]  = (bf16_t)y0.x; a[9]  = (bf16_t)y0.y; a[10] = (bf16_t)y0.z; a[11] = (bf16_t)y0.w;
  a[12] = (bf16_t)y1.x; a[13] = (bf16_t)y1.y; a[14] = (bf16_t)y1.z; a[15] = (bf16_t)y1.w;
  return a;
}

__device__ __forceinline__ v16bf load_a_frag_bf16(const bf16_t* __restrict__ Arow,
                                                  int kbase, int half) {
  const v8bf r0 = *(const v8bf*)(Arow + kbase + half * 8);
  const v8bf r1 = *(const v8bf*)(Arow + kbase + 16 + half * 8);
  v16bf a;
#pragma unroll
  for (int i = 0; i < 8; ++i) { a[i] = r0[i]; a[8 + i] = r1[i]; }
  return a;
}

#define WMMA_BF16(A, B, C) \
  __builtin_amdgcn_wmma_f32_16x16x32_bf16(false, (A), false, (B), (short)0, (C), false, false)

// ---------------------------------------------------------------------
// WMMA GEMM:  C[M,Ncol] = act( cat(A0[M,K0], A1[M,K1]) @ B + bias )
// B pre-transposed/converted: Bt[Ncol][Ktot] bf16 (lane B-fragment = two
// contiguous 16B runs). 8 waves/block; each wave: 16(M) x 64(N), 4 accums
// sharing one A fragment -> 4 back-to-back v_wmma per K-step (4x A reuse).
// Branchless K-loop: A row index clamped (garbage rows never stored).
// ---------------------------------------------------------------------
template<int ACT>
__global__ void gemm_wmma_bf16(const float* __restrict__ A0,
                               const float* __restrict__ A1,
                               int K0, int Ktot,
                               const bf16_t* __restrict__ Bt,
                               const float* __restrict__ bias,
                               float* __restrict__ C,
                               int M, int Ncol) {
  const int lane  = threadIdx.x & 31;
  const int wave  = threadIdx.x >> 5;
  const int half  = lane >> 4;
  const int mr    = lane & 15;
  const int tileM = blockIdx.x * 128 + wave * 16;
  const int tileN = blockIdx.y * 64;
  const int row   = min(tileM + mr, M - 1);     // clamp: no K-loop branches
  const int K1    = Ktot - K0;

  const float* __restrict__ Arow0 = A0 + (size_t)row * K0;
  const float* __restrict__ Arow1 = A1 ? (A1 + (size_t)row * K1) : nullptr;
  int cols[4];
  const bf16_t* Brow[4];
#pragma unroll
  for (int i = 0; i < 4; ++i) {
    cols[i] = tileN + 16 * i + mr;
    Brow[i] = Bt + (size_t)cols[i] * Ktot;
  }

  v8f acc0 = {}, acc1 = {}, acc2 = {}, acc3 = {};
  for (int kb = 0; kb < Ktot; kb += 32) {
    // K0 is a multiple of 32 -> each 32-wide K block is within one source
    const v16bf a = (kb < K0) ? load_a_frag_f32(Arow0, kb, half)
                              : load_a_frag_f32(Arow1, kb - K0, half);
    // B fragment: same K order as A -> runs [h*8, h*8+8) and [16+h*8, 16+h*8+8)
    const v16bf b0 = load_a_frag_bf16(Brow[0], kb, half);
    const v16bf b1 = load_a_frag_bf16(Brow[1], kb, half);
    const v16bf b2 = load_a_frag_bf16(Brow[2], kb, half);
    const v16bf b3 = load_a_frag_bf16(Brow[3], kb, half);
    acc0 = WMMA_BF16(a, b0, acc0);
    acc1 = WMMA_BF16(a, b1, acc1);
    acc2 = WMMA_BF16(a, b2, acc2);
    acc3 = WMMA_BF16(a, b3, acc3);
  }
  // C/D layout: lanes 0-15: M=v, N=lane ; lanes 16-31: M=v+8, N=lane-16
  float bia[4] = {0.f, 0.f, 0.f, 0.f};
  if (bias) {
#pragma unroll
    for (int i = 0; i < 4; ++i) bia[i] = bias[cols[i]];
  }
#pragma unroll
  for (int v = 0; v < 8; ++v) {
    const int m = tileM + v + (half ? 8 : 0);
    if (m < M) {
      float vv[4] = {acc0[v] + bia[0], acc1[v] + bia[1],
                     acc2[v] + bia[2], acc3[v] + bia[3]};
#pragma unroll
      for (int i = 0; i < 4; ++i) {
        if (ACT == ACT_RELU)          vv[i] = fmaxf(vv[i], 0.f);
        else if (ACT == ACT_SOFTPLUS) vv[i] = softplusf_(vv[i]);
        C[(size_t)m * Ncol + cols[i]] = vv[i];
      }
    }
  }
}

// ---------------------------------------------------------------------
// Fused inner-product decoder + weighted BCE reduction.
// logits tile = z @ z^T via WMMA on pre-converted bf16 z (never
// materialized; adj streamed exactly once). 8 waves/block, each wave
// 16x64 (4 accums), block tile 64x128.
// ---------------------------------------------------------------------
__global__ void decoder_bce_wmma(const bf16_t* __restrict__ zbf,
                                 const float* __restrict__ adj,
                                 const float* __restrict__ adjsum,
                                 float* __restrict__ bce_accum,
                                 int n, int zdim) {
  const int lane = threadIdx.x & 31;
  const int wave = threadIdx.x >> 5;
  const int wm = wave & 3, wn = wave >> 2;
  const int half = lane >> 4, mr = lane & 15;
  const int tileM = blockIdx.x * 64 + wm * 16;
  const int tileN = blockIdx.y * 128 + wn * 64;
  const int rowA = min(tileM + mr, n - 1);

  int cols[4];
  const bf16_t* Br[4];
#pragma unroll
  for (int i = 0; i < 4; ++i) {
    cols[i] = tileN + 16 * i + mr;                       // logical col
    Br[i] = zbf + (size_t)min(cols[i], n - 1) * zdim;    // clamped z row
  }
  const bf16_t* __restrict__ Ar = zbf + (size_t)rowA * zdim;

  v8f acc0 = {}, acc1 = {}, acc2 = {}, acc3 = {};
  for (int kb = 0; kb < zdim; kb += 32) {
    const v16bf a  = load_a_frag_bf16(Ar, kb, half);
    const v16bf b0 = load_a_frag_bf16(Br[0], kb, half);
    const v16bf b1 = load_a_frag_bf16(Br[1], kb, half);
    const v16bf b2 = load_a_frag_bf16(Br[2], kb, half);
    const v16bf b3 = load_a_frag_bf16(Br[3], kb, half);
    acc0 = WMMA_BF16(a, b0, acc0);
    acc1 = WMMA_BF16(a, b1, acc1);
    acc2 = WMMA_BF16(a, b2, acc2);
    acc3 = WMMA_BF16(a, b3, acc3);
  }

  const float s    = *adjsum;
  const float nn   = (float)n * (float)n;
  const float posw = (nn - s) / s;
  float local = 0.f;
#pragma unroll
  for (int v = 0; v < 8; ++v) {
    const int m = tileM + v + (half ? 8 : 0);
    if (m < n) {
      const float* arow = adj + (size_t)m * n;
      const float lg[4] = {acc0[v], acc1[v], acc2[v], acc3[v]};
#pragma unroll
      for (int i = 0; i < 4; ++i) {
        if (cols[i] < n) {
          const float l = lg[i], t = arow[cols[i]];
          local += posw * t * softplusf_(-l) + (1.f - t) * softplusf_(l);
        }
      }
    }
  }
  __shared__ float red[256];
  red[threadIdx.x] = local;
  __syncthreads();
  for (int off = 128; off > 0; off >>= 1) {
    if (threadIdx.x < off) red[threadIdx.x] += red[threadIdx.x + off];
    __syncthreads();
  }
  if (threadIdx.x == 0) atomicAdd(bce_accum, red[0]);
}

// ------------------- weight / activation conversion -------------------
// Wt[n][k] = (bf16) W[k][n]
__global__ void w_to_bf16t(const float* __restrict__ W, bf16_t* __restrict__ Wt,
                           int K, int Ncol) {
  const long long i = (long long)blockIdx.x * blockDim.x + threadIdx.x;
  if (i < (long long)K * Ncol) {
    const int k = (int)(i / Ncol), n = (int)(i % Ncol);
    Wt[(size_t)n * K + k] = (bf16_t)W[i];
  }
}
__global__ void f32_to_bf16(const float* __restrict__ s, bf16_t* __restrict__ d,
                            long long n) {
  const long long i = (long long)blockIdx.x * blockDim.x + threadIdx.x;
  if (i < n) d[i] = (bf16_t)s[i];
}

// ---------------------------- GCN support ----------------------------
__global__ void deg_scatter(const int* __restrict__ row, int E, int n,
                            float* __restrict__ deg) {
  const int i = blockIdx.x * blockDim.x + threadIdx.x;
  if (i < E)          atomicAdd(&deg[row[i]], 1.f);
  else if (i < E + n) atomicAdd(&deg[i - E], 2.f);   // improved self-loop w=2
}

__global__ void deg_invsqrt(float* __restrict__ deg, int n) {
  const int i = blockIdx.x * blockDim.x + threadIdx.x;
  if (i < n) { const float d = deg[i]; deg[i] = (d > 0.f) ? rsqrtf(d) : 0.f; }
}

// dst[r] += dinv[r]*w*dinv[c] * src[c]   (float4 per thread)
// f4s = log2(F/4): F is a power of two -> shift/mask instead of div/mod
__global__ void conv_scatter(const int* __restrict__ row, const int* __restrict__ col,
                             int E, int n, const float* __restrict__ dinv,
                             const float* __restrict__ src, float* __restrict__ dst,
                             int F, int f4s) {
  const long long idx = (long long)blockIdx.x * blockDim.x + threadIdx.x;
  const long long tot = (long long)(E + n) << f4s;
  if (idx >= tot) return;
  const int e = (int)(idx >> f4s);
  const int f = ((int)idx & ((1 << f4s) - 1)) * 4;
  int r, c; float w;
  if (e < E) { r = row[e]; c = col[e]; w = 1.f; }
  else       { r = c = e - E;          w = 2.f; }
  const float nrm = dinv[r] * w * dinv[c];
  const float4 sv = *(const float4*)(src + (size_t)c * F + f);
  float* d = dst + (size_t)r * F + f;
  atomicAdd(d + 0, nrm * sv.x);
  atomicAdd(d + 1, nrm * sv.y);
  atomicAdd(d + 2, nrm * sv.z);
  atomicAdd(d + 3, nrm * sv.w);
}

// ---------------------------- elementwise ----------------------------
__global__ void fill_f(float* __restrict__ p, float v, long long n) {
  const long long i = (long long)blockIdx.x * blockDim.x + threadIdx.x;
  if (i < n) p[i] = v;
}
__global__ void copy_f(float* __restrict__ d, const float* __restrict__ s, long long n) {
  const long long i = (long long)blockIdx.x * blockDim.x + threadIdx.x;
  if (i < n) d[i] = s[i];
}
__global__ void ew_relu(float* __restrict__ a, long long n) {
  const long long i = (long long)blockIdx.x * blockDim.x + threadIdx.x;
  if (i < n) a[i] = fmaxf(a[i], 0.f);
}
__global__ void ew_softplus(float* __restrict__ a, long long n) {
  const long long i = (long long)blockIdx.x * blockDim.x + threadIdx.x;
  if (i < n) a[i] = softplusf_(a[i]);
}
__global__ void ew_sigmoid(float* __restrict__ a, long long n) {
  const long long i = (long long)blockIdx.x * blockDim.x + threadIdx.x;
  if (i < n) a[i] = 1.f / (1.f + expf(-a[i]));
}
__global__ void ew_zcombine(float* __restrict__ z, const float* __restrict__ m,
                            const float* __restrict__ ep, const float* __restrict__ sd,
                            long long n) {
  const long long i = (long long)blockIdx.x * blockDim.x + threadIdx.x;
  if (i < n) z[i] = m[i] + ep[i] * sd[i];
}
__global__ void ew_mul(float* __restrict__ o, const float* __restrict__ a,
                       const float* __restrict__ b, long long n) {
  const long long i = (long long)blockIdx.x * blockDim.x + threadIdx.x;
  if (i < n) o[i] = a[i] * b[i];
}
__global__ void ew_hupdate(float* __restrict__ h, const float* __restrict__ g,
                           const float* __restrict__ c, long long n) {
  const long long i = (long long)blockIdx.x * blockDim.x + threadIdx.x;
  if (i < n) { const float gg = g[i]; h[i] = gg * h[i] + (1.f - gg) * tanhf(c[i]); }
}

// ---------------------------- reductions -----------------------------
__global__ void kld_reduce(const float* __restrict__ m1, const float* __restrict__ s1,
                           const float* __restrict__ m2, const float* __restrict__ s2,
                           float* __restrict__ accum, long long count) {
  float local = 0.f;
  for (long long i = (long long)blockIdx.x * blockDim.x + threadIdx.x; i < count;
       i += (long long)gridDim.x * blockDim.x) {
    const float a = s1[i] + EPSF, b = s2[i] + EPSF;
    const float dm = m1[i] - m2[i];
    local += 2.f * logf(b) - 2.f * logf(a) + (a * a + dm * dm) / (b * b) - 1.f;
  }
  __shared__ float red[256];
  red[threadIdx.x] = local;
  __syncthreads();
  for (int off = 128; off > 0; off >>= 1) {
    if (threadIdx.x < off) red[threadIdx.x] += red[threadIdx.x + off];
    __syncthreads();
  }
  if (threadIdx.x == 0) atomicAdd(accum, red[0]);
}

__global__ void sum_reduce(const float* __restrict__ a, float* __restrict__ accum,
                           long long count) {
  float local = 0.f;
  for (long long i = (long long)blockIdx.x * blockDim.x + threadIdx.x; i < count;
       i += (long long)gridDim.x * blockDim.x) local += a[i];
  __shared__ float red[256];
  red[threadIdx.x] = local;
  __syncthreads();
  for (int off = 128; off > 0; off >>= 1) {
    if (threadIdx.x < off) red[threadIdx.x] += red[threadIdx.x + off];
    __syncthreads();
  }
  if (threadIdx.x == 0) atomicAdd(accum, red[0]);
}

// scal[0]=kld_sum, scal[1]=bce_sum, scal[2]=adj_sum
__global__ void finalize_t(const float* __restrict__ scal, float* __restrict__ out) {
  if (threadIdx.x == 0 && blockIdx.x == 0) {
    const float nfl = (float)NN_;
    const float nn  = nfl * nfl;
    out[0] += 0.5f * scal[0] / nn;                 // (0.5/N) * mean(rowsum)
    const float s    = scal[2];
    const float norm = nn / ((nn - s) * 2.f);
    out[1] += norm * (scal[1] / nn);               // norm * mean(bce)
  }
}

// ---------------------------------------------------------------------
static inline int blks(long long n, int b) { return (int)((n + b - 1) / b); }

extern "C" void kernel_launch(void* const* d_in, const int* in_sizes, int n_in,
                              void* d_out, int out_size, void* d_ws, size_t ws_size,
                              hipStream_t stream) {
  const float* x       = (const float*)d_in[0];
  const int*   eidx    = (const int*)  d_in[1];
  const float* adj     = (const float*)d_in[2];
  const float* eps_z   = (const float*)d_in[3];
  const float* W_phix  = (const float*)d_in[4];
  const float* W_phiz  = (const float*)d_in[5];
  const float* W_enc   = (const float*)d_in[6];
  const float* W_encm  = (const float*)d_in[7];
  const float* W_encs  = (const float*)d_in[8];
  const float* W_pri   = (const float*)d_in[9];
  const float* b_pri   = (const float*)d_in[10];
  const float* W_prim  = (const float*)d_in[11];
  const float* b_prim  = (const float*)d_in[12];
  const float* W_pris  = (const float*)d_in[13];
  const float* b_pris  = (const float*)d_in[14];
  const float* W_xz    = (const float*)d_in[15];
  const float* W_hz    = (const float*)d_in[16];
  const float* W_xr    = (const float*)d_in[17];
  const float* W_hr    = (const float*)d_in[18];
  const float* W_xh    = (const float*)d_in[19];
  const float* W_hh    = (const float*)d_in[20];

  float* out = (float*)d_out;
  const size_t NH = (size_t)NN_ * HD_;
  const size_t NZ = (size_t)NN_ * ZD_;
  float* out_encm = out + 2;                         // [T,N,Z]
  float* out_prim = out + 2 + (size_t)TT_ * NZ;      // [T,N,Z]
  float* out_h    = out + 2 + 2 * (size_t)TT_ * NZ;  // [N,H]

  // -------- workspace carve-up --------
  float* ws = (float*)d_ws;
  float* phi_x   = ws;  ws += NH;
  float* encb    = ws;  ws += NH;
  float* tmp     = ws;  ws += NH;
  float* gz      = ws;  ws += NH;
  float* gr      = ws;  ws += NH;
  float* gc      = ws;  ws += NH;
  float* rh      = ws;  ws += NH;
  float* prior   = ws;  ws += NH;
  float* phi_z   = ws;  ws += NH;
  float* hbuf    = ws;  ws += NH;
  float* enc_std = ws;  ws += NZ;
  float* pri_std = ws;  ws += NZ;
  float* zbuf    = ws;  ws += NZ;
  float* deg     = ws;  ws += NN_;
  float* scal    = ws;  ws += 8;     // [kld_sum, bce_sum, adj_sum]
  // bf16 region
  bf16_t* bw = (bf16_t*)ws;
  bf16_t* T_phix = bw;  bw += (size_t)XD_ * HD_;       // [256][256]
  bf16_t* T_phiz = bw;  bw += (size_t)HD_ * ZD_;       // [256][128]
  bf16_t* T_enc  = bw;  bw += (size_t)HD_ * 2 * HD_;   // [256][512]
  bf16_t* T_encm = bw;  bw += (size_t)ZD_ * HD_;       // [128][256]
  bf16_t* T_encs = bw;  bw += (size_t)ZD_ * HD_;
  bf16_t* T_pri  = bw;  bw += (size_t)HD_ * HD_;
  bf16_t* T_prim = bw;  bw += (size_t)ZD_ * HD_;
  bf16_t* T_pris = bw;  bw += (size_t)ZD_ * HD_;
  bf16_t* T_xz   = bw;  bw += (size_t)HD_ * 2 * HD_;
  bf16_t* T_hz   = bw;  bw += (size_t)HD_ * HD_;
  bf16_t* T_xr   = bw;  bw += (size_t)HD_ * 2 * HD_;
  bf16_t* T_hr   = bw;  bw += (size_t)HD_ * HD_;
  bf16_t* T_xh   = bw;  bw += (size_t)HD_ * 2 * HD_;
  bf16_t* T_hh   = bw;  bw += (size_t)HD_ * HD_;
  bf16_t* zbf    = bw;  bw += NZ;

  const int TB = 256;
  const dim3 bgemm(256);
  const dim3 g256(blks(NN_, 128), HD_ / 64);   // Ncol=256 -> grid.y=4
  const dim3 g128(blks(NN_, 128), ZD_ / 64);   // Ncol=128 -> grid.y=2
  const dim3 gdec(blks(NN_, 64), blks(NN_, 128)), bdec(256);

  // -------- one-time (per launch) weight transpose+convert to bf16 --------
  {
    struct { const float* w; bf16_t* t; int K, Nc; } cv[14] = {
      {W_phix, T_phix, XD_, HD_},     {W_phiz, T_phiz, ZD_, HD_},
      {W_enc,  T_enc,  2*HD_, HD_},   {W_encm, T_encm, HD_, ZD_},
      {W_encs, T_encs, HD_, ZD_},     {W_pri,  T_pri,  HD_, HD_},
      {W_prim, T_prim, HD_, ZD_},     {W_pris, T_pris, HD_, ZD_},
      {W_xz,   T_xz,   2*HD_, HD_},   {W_hz,   T_hz,   HD_, HD_},
      {W_xr,   T_xr,   2*HD_, HD_},   {W_hr,   T_hr,   HD_, HD_},
      {W_xh,   T_xh,   2*HD_, HD_},   {W_hh,   T_hh,   HD_, HD_},
    };
    for (int i = 0; i < 14; ++i)
      w_to_bf16t<<<blks((long long)cv[i].K * cv[i].Nc, TB), TB, 0, stream>>>(
          cv[i].w, cv[i].t, cv[i].K, cv[i].Nc);
  }

  // init: h = 0, out[0..1] = 0
  fill_f<<<blks(NH, TB), TB, 0, stream>>>(hbuf, 0.f, (long long)NH);
  fill_f<<<1, 32, 0, stream>>>(out, 0.f, 2);

  for (int t = 0; t < TT_; ++t) {
    const int*   row   = eidx + (size_t)t * 2 * EE_;
    const int*   col   = row + EE_;
    const float* x_t   = x     + (size_t)t * NN_ * XD_;
    const float* adj_t = adj   + (size_t)t * NN_ * NN_;
    const float* eps_t = eps_z + (size_t)t * NZ;
    float* encm_t = out_encm + (size_t)t * NZ;
    float* prim_t = out_prim + (size_t)t * NZ;

    fill_f<<<1, 32, 0, stream>>>(scal, 0.f, 3);

    // degree -> dinv (improved self-loops, weight 2)
    fill_f<<<blks(NN_, TB), TB, 0, stream>>>(deg, 0.f, NN_);
    deg_scatter<<<blks(EE_ + NN_, TB), TB, 0, stream>>>(row, EE_, NN_, deg);
    deg_invsqrt<<<blks(NN_, TB), TB, 0, stream>>>(deg, NN_);

    // phi_x = relu(x @ W_phix)
    gemm_wmma_bf16<ACT_RELU><<<g256, bgemm, 0, stream>>>(
        x_t, nullptr, XD_, XD_, T_phix, nullptr, phi_x, NN_, HD_);

    // enc = relu(conv(cat(phi_x, h) @ W_enc))
    gemm_wmma_bf16<ACT_NONE><<<g256, bgemm, 0, stream>>>(
        phi_x, hbuf, HD_, 2 * HD_, T_enc, nullptr, tmp, NN_, HD_);
    fill_f<<<blks(NH, TB), TB, 0, stream>>>(encb, 0.f, (long long)NH);
    conv_scatter<<<blks((long long)(EE_ + NN_) * (HD_ / 4), TB), TB, 0, stream>>>(
        row, col, EE_, NN_, deg, tmp, encb, HD_, 6);
    ew_relu<<<blks(NH, TB), TB, 0, stream>>>(encb, (long long)NH);

    // enc_mean = conv(enc @ W_encm) -> directly into d_out slice
    gemm_wmma_bf16<ACT_NONE><<<g128, bgemm, 0, stream>>>(
        encb, nullptr, HD_, HD_, T_encm, nullptr, tmp, NN_, ZD_);
    fill_f<<<blks(NZ, TB), TB, 0, stream>>>(encm_t, 0.f, (long long)NZ);
    conv_scatter<<<blks((long long)(EE_ + NN_) * (ZD_ / 4), TB), TB, 0, stream>>>(
        row, col, EE_, NN_, deg, tmp, encm_t, ZD_, 5);

    // enc_std = softplus(conv(enc @ W_encs))
    gemm_wmma_bf16<ACT_NONE><<<g128, bgemm, 0, stream>>>(
        encb, nullptr, HD_, HD_, T_encs, nullptr, tmp, NN_, ZD_);
    fill_f<<<blks(NZ, TB), TB, 0, stream>>>(enc_std, 0.f, (long long)NZ);
    conv_scatter<<<blks((long long)(EE_ + NN_) * (ZD_ / 4), TB), TB, 0, stream>>>(
        row, col, EE_, NN_, deg, tmp, enc_std, ZD_, 5);
    ew_softplus<<<blks(NZ, TB), TB, 0, stream>>>(enc_std, (long long)NZ);

    // prior MLP
    gemm_wmma_bf16<ACT_RELU><<<g256, bgemm, 0, stream>>>(
        hbuf, nullptr, HD_, HD_, T_pri, b_pri, prior, NN_, HD_);
    gemm_wmma_bf16<ACT_NONE><<<g128, bgemm, 0, stream>>>(
        prior, nullptr, HD_, HD_, T_prim, b_prim, prim_t, NN_, ZD_);
    gemm_wmma_bf16<ACT_SOFTPLUS><<<g128, bgemm, 0, stream>>>(
        prior, nullptr, HD_, HD_, T_pris, b_pris, pri_std, NN_, ZD_);

    // z = enc_mean + eps * enc_std ; phi_z = relu(z @ W_phiz)
    ew_zcombine<<<blks(NZ, TB), TB, 0, stream>>>(zbuf, encm_t, eps_t, enc_std,
                                                 (long long)NZ);
    gemm_wmma_bf16<ACT_RELU><<<g256, bgemm, 0, stream>>>(
        zbuf, nullptr, ZD_, ZD_, T_phiz, nullptr, phi_z, NN_, HD_);

    // GRU: zg
    fill_f<<<blks(NH, TB), TB, 0, stream>>>(gz, 0.f, (long long)NH);
    gemm_wmma_bf16<ACT_NONE><<<g256, bgemm, 0, stream>>>(
        phi_x, phi_z, HD_, 2 * HD_, T_xz, nullptr, tmp, NN_, HD_);
    conv_scatter<<<blks((long long)(EE_ + NN_) * (HD_ / 4), TB), TB, 0, stream>>>(
        row, col, EE_, NN_, deg, tmp, gz, HD_, 6);
    gemm_wmma_bf16<ACT_NONE><<<g256, bgemm, 0, stream>>>(
        hbuf, nullptr, HD_, HD_, T_hz, nullptr, tmp, NN_, HD_);
    conv_scatter<<<blks((long long)(EE_ + NN_) * (HD_ / 4), TB), TB, 0, stream>>>(
        row, col, EE_, NN_, deg, tmp, gz, HD_, 6);
    ew_sigmoid<<<blks(NH, TB), TB, 0, stream>>>(gz, (long long)NH);

    // rg
    fill_f<<<blks(NH, TB), TB, 0, stream>>>(gr, 0.f, (long long)NH);
    gemm_wmma_bf16<ACT_NONE><<<g256, bgemm, 0, stream>>>(
        phi_x, phi_z, HD_, 2 * HD_, T_xr, nullptr, tmp, NN_, HD_);
    conv_scatter<<<blks((long long)(EE_ + NN_) * (HD_ / 4), TB), TB, 0, stream>>>(
        row, col, EE_, NN_, deg, tmp, gr, HD_, 6);
    gemm_wmma_bf16<ACT_NONE><<<g256, bgemm, 0, stream>>>(
        hbuf, nullptr, HD_, HD_, T_hr, nullptr, tmp, NN_, HD_);
    conv_scatter<<<blks((long long)(EE_ + NN_) * (HD_ / 4), TB), TB, 0, stream>>>(
        row, col, EE_, NN_, deg, tmp, gr, HD_, 6);
    ew_sigmoid<<<blks(NH, TB), TB, 0, stream>>>(gr, (long long)NH);

    // h_tilde pre-activation
    ew_mul<<<blks(NH, TB), TB, 0, stream>>>(rh, gr, hbuf, (long long)NH);
    fill_f<<<blks(NH, TB), TB, 0, stream>>>(gc, 0.f, (long long)NH);
    gemm_wmma_bf16<ACT_NONE><<<g256, bgemm, 0, stream>>>(
        phi_x, phi_z, HD_, 2 * HD_, T_xh, nullptr, tmp, NN_, HD_);
    conv_scatter<<<blks((long long)(EE_ + NN_) * (HD_ / 4), TB), TB, 0, stream>>>(
        row, col, EE_, NN_, deg, tmp, gc, HD_, 6);
    gemm_wmma_bf16<ACT_NONE><<<g256, bgemm, 0, stream>>>(
        rh, nullptr, HD_, HD_, T_hh, nullptr, tmp, NN_, HD_);
    conv_scatter<<<blks((long long)(EE_ + NN_) * (HD_ / 4), TB), TB, 0, stream>>>(
        row, col, EE_, NN_, deg, tmp, gc, HD_, 6);

    // h = zg*h + (1-zg)*tanh(gc)
    ew_hupdate<<<blks(NH, TB), TB, 0, stream>>>(hbuf, gz, gc, (long long)NH);

    // KLD partial sum
    kld_reduce<<<1024, 256, 0, stream>>>(encm_t, enc_std, prim_t, pri_std,
                                         &scal[0], (long long)NZ);
    // adj sum (posw), z->bf16, fused z z^T + BCE
    sum_reduce<<<2048, 256, 0, stream>>>(adj_t, &scal[2], (long long)NN_ * NN_);
    f32_to_bf16<<<blks(NZ, TB), TB, 0, stream>>>(zbuf, zbf, (long long)NZ);
    decoder_bce_wmma<<<gdec, bdec, 0, stream>>>(zbf, adj_t, &scal[2], &scal[1],
                                                NN_, ZD_);
    finalize_t<<<1, 32, 0, stream>>>(scal, out);
  }

  // final hidden state
  copy_f<<<blks(NH, TB), TB, 0, stream>>>(out_h, hbuf, (long long)NH);
}
